// EquiformerV2Block_84859963834911
// MI455X (gfx1250) — compile-verified, matrix operand order checked by reading, not run on hardware
//
#include <hip/hip_runtime.h>
#include <hip/hip_bf16.h>
#include <math.h>

// ---------------------------------------------------------------------------
// EquiformerV2 block for MI455X (gfx1250, wave32, WMMA).
// - GEMMs: bf16 WMMA (v_wmma_f32_16x16x32_bf16, f32 accum), double-buffered
//   LDS tiles staged in fragment order (operands = 2x ds_load_b128 each),
//   branchless vectorized global staging (global_load_b128 + cndmask).
// - Per-edge staging uses CDNA5 async global->LDS copies
//   (global_load_async_to_lds_b128 + s_wait_asynccnt).
// ---------------------------------------------------------------------------

#define L2C 25
#define M2C 19
#define CCH 64
#define NRBF 600
#define DIN 728          // NRBF + 2*CE
#define HEADS 8
#define VCH 16
#define HMSG 64
#define HFF 128
#define GRD 288
#define NLAY 2

typedef unsigned short u16;
typedef __attribute__((ext_vector_type(16))) __bf16 v16bf;
typedef __attribute__((ext_vector_type(8)))  float v8f;

__constant__ int c_LOC[25] = {0, 1,1,1, 2,2,2,2,2, 3,3,3,3,3,3,3, 4,4,4,4,4,4,4,4,4};
__constant__ int c_MG[19]  = {0, 1,0,1, 2,1,0,1,2, 2,1,0,1,2, 2,1,0,1,2};

__device__ __forceinline__ float silu_f(float x) { return x / (1.0f + __expf(-x)); }

__device__ __forceinline__ u16 f2bf_bits(float f) {
    unsigned u = __float_as_uint(f);
    unsigned r = u + 0x7FFFu + ((u >> 16) & 1u);
    return (u16)(r >> 16);
}
__device__ __forceinline__ unsigned pack2(float a, float b) {
    return (unsigned)f2bf_bits(a) | ((unsigned)f2bf_bits(b) << 16);
}

__device__ __forceinline__ unsigned fenc(float f) {
    unsigned u = __float_as_uint(f);
    return (u & 0x80000000u) ? ~u : (u | 0x80000000u);
}
__device__ __forceinline__ float fdec(unsigned u) {
    u = (u & 0x80000000u) ? (u & 0x7FFFFFFFu) : ~u;
    return __uint_as_float(u);
}

// ---------------------------------------------------------------------------
// CDNA5 async global->LDS copy (ASYNCcnt-tracked). The LDS VDST operand is
// the LDS byte offset = low 32 bits of the flat shared-pointer.
// ---------------------------------------------------------------------------
__device__ __forceinline__ void g2l_b128(void* lds_dst, const void* gsrc) {
    unsigned loff = (unsigned)(size_t)lds_dst;
    asm volatile("global_load_async_to_lds_b128 %0, %1, off"
                 :: "v"(loff), "v"(gsrc) : "memory");
}
__device__ __forceinline__ void wait_async0() {
    asm volatile("s_wait_asynccnt 0x0" ::: "memory");
}
// nfloat must be a multiple of 4
__device__ __forceinline__ void lds_copy_async(float* dst, const float* src, int nfloat) {
    for (int t = threadIdx.x; t * 4 < nfloat; t += blockDim.x)
        g2l_b128(dst + t * 4, src + t * 4);
    wait_async0();
}

// ---------------------------------------------------------------------------
// WMMA GEMM:  C[M,N] = act( A[M,K](lda) @ W[K,N](ldw) + bias )
// Block tile 128x64, BK=32, 256 threads = 8 waves (4 M x 2 N), each wave a
// 32x32 patch via 2x2 v_wmma_f32_16x16x32_bf16. Double-buffered LDS tiles.
// Fragment element j of lane L maps to tile (row/col = subtile*16 + (L&15),
// k = (j&7) + 8*((L>>4)&1) + 16*(j>>3)); [subtile][lane][24] layout keeps
// 16 B alignment so every operand is two ds_load_b128.
// Requires K % 8 == 0 (true for all call sites: 728/128/64) so each k-octet
// is fully in- or out-of-range -> branchless clamped loads + masking.
// ---------------------------------------------------------------------------
template<int AIN_SILU, int AOUT_SILU>
__global__ __launch_bounds__(256)
void k_gemm(const float* __restrict__ A, int lda,
            const float* __restrict__ W, int ldw,
            const float* __restrict__ bias,
            float* __restrict__ C, int ldc,
            int M, int N, int K)
{
    __shared__ u16 As2[2][8][32][24];   // ping-pong, 8 M-subtiles of 16x32
    __shared__ u16 Bs2[2][4][32][24];   // ping-pong, 4 N-subtiles of 32x16

    const int tid  = threadIdx.x;
    const int row0 = blockIdx.x * 128;
    const int col0 = blockIdx.y * 64;

    const int lane = tid & 31;
    const int wave = tid >> 5;
    const int wm   = wave & 3;
    const int wn   = wave >> 2;
    const int r    = lane & 15;
    const int hi   = lane >> 4;

    v8f acc[2][2] = {};

    auto stage = [&](int buf, int k0) {
        // --- A tile (128x32): 512 octets of 8 consecutive k ---
        #pragma unroll
        for (int g = tid; g < 512; g += 256) {
            int rr = g >> 2, q = g & 3;
            int gr = row0 + rr;
            int kb = k0 + q * 8;
            bool ok = (gr < M) & (kb + 8 <= K);
            const float4* s4 =
                (const float4*)(A + (size_t)(ok ? gr : 0) * lda + (ok ? kb : 0));
            float4 a0 = s4[0], a1 = s4[1];
            float f[8] = {a0.x, a0.y, a0.z, a0.w, a1.x, a1.y, a1.z, a1.w};
            #pragma unroll
            for (int i = 0; i < 8; ++i) {
                float v = ok ? f[i] : 0.0f;
                if (AIN_SILU) v = silu_f(v);   // silu(0)==0 keeps padding zero
                f[i] = v;
            }
            uint4 pk;
            pk.x = pack2(f[0], f[1]); pk.y = pack2(f[2], f[3]);
            pk.z = pack2(f[4], f[5]); pk.w = pack2(f[6], f[7]);
            *(uint4*)&As2[buf][rr >> 4][(rr & 15) + ((q & 1) << 4)][(q >> 1) * 8] = pk;
        }
        // --- B tile (32x64): one k-octet column per thread ---
        {
            int cc = tid & 63, q = tid >> 6;
            int gc = col0 + cc;
            int kb = k0 + q * 8;
            bool ok = (gc < N) & (kb + 8 <= K);
            int gcc = ok ? gc : 0;
            int kbc = ok ? kb : 0;
            float f[8];
            #pragma unroll
            for (int i = 0; i < 8; ++i) f[i] = W[(size_t)(kbc + i) * ldw + gcc];
            #pragma unroll
            for (int i = 0; i < 8; ++i) f[i] = ok ? f[i] : 0.0f;
            uint4 pk;
            pk.x = pack2(f[0], f[1]); pk.y = pack2(f[2], f[3]);
            pk.z = pack2(f[4], f[5]); pk.w = pack2(f[6], f[7]);
            *(uint4*)&Bs2[buf][cc >> 4][(cc & 15) + ((q & 1) << 4)][(q >> 1) * 8] = pk;
        }
        // prefetch one more tile ahead (global_prefetch_b8)
        if (k0 + 32 < K) {
            int pr = row0 + (tid >> 1);
            if (pr < M) __builtin_prefetch(A + (size_t)pr * lda + k0 + 32, 0, 1);
        }
    };

    const int KT = (K + 31) / 32;
    stage(0, 0);
    for (int kt = 0; kt < KT; ++kt) {
        const int cur = kt & 1;
        __syncthreads();                         // buffer `cur` ready; prior reads done
        if (kt + 1 < KT) stage(cur ^ 1, (kt + 1) * 32);

        union FragU { uint4 q2[2]; v16bf b; };
        FragU af[2], bfr[2];
        #pragma unroll
        for (int mi = 0; mi < 2; ++mi) {
            const u16* p = &As2[cur][wm * 2 + mi][lane][0];
            af[mi].q2[0] = *(const uint4*)p;
            af[mi].q2[1] = *(const uint4*)(p + 8);
        }
        #pragma unroll
        for (int ni = 0; ni < 2; ++ni) {
            const u16* p = &Bs2[cur][wn * 2 + ni][lane][0];
            bfr[ni].q2[0] = *(const uint4*)p;
            bfr[ni].q2[1] = *(const uint4*)(p + 8);
        }
        #pragma unroll
        for (int mi = 0; mi < 2; ++mi)
            #pragma unroll
            for (int ni = 0; ni < 2; ++ni)
                acc[mi][ni] = __builtin_amdgcn_wmma_f32_16x16x32_bf16(
                    false, af[mi].b, false, bfr[ni].b,
                    (short)0, acc[mi][ni], false, false);
    }

    // epilogue: C/D layout -> lane (N=lane&15, half=lane>>4), VGPR v = M row
    const bool full = (row0 + 128 <= M) && (col0 + 64 <= N);
    #pragma unroll
    for (int mi = 0; mi < 2; ++mi) {
        #pragma unroll
        for (int ni = 0; ni < 2; ++ni) {
            int col = col0 + wn * 32 + ni * 16 + r;
            float b = (bias && (full || col < N)) ? bias[col < N ? col : 0] : 0.0f;
            #pragma unroll
            for (int v = 0; v < 8; ++v) {
                int row = row0 + wm * 32 + mi * 16 + v + 8 * hi;
                if (full || (row < M && col < N)) {
                    float val = acc[mi][ni][v] + b;
                    if (AOUT_SILU) val = silu_f(val);
                    C[(size_t)row * ldc + col] = val;
                }
            }
        }
    }
}

// ---------------------------------------------------------------------------
// Elementwise / per-edge kernels
// ---------------------------------------------------------------------------

__global__ __launch_bounds__(256)
void k_ee(const float* __restrict__ ev, const float* __restrict__ ese,
          const int* __restrict__ snd, const int* __restrict__ rcv,
          const int* __restrict__ nspec, float* __restrict__ ee, int E)
{
    int e = blockIdx.x;
    if (e >= E) return;
    float x = ev[(size_t)e * 3 + 0], y = ev[(size_t)e * 3 + 1], z = ev[(size_t)e * 3 + 2];
    float d = sqrtf(x * x + y * y + z * z + 1e-12f);
    const float sp = 5.0f / 599.0f;
    const float inv_std = 1.0f / (2.0f * sp);
    int ss = nspec[snd[e]];
    int rs = nspec[rcv[e]];
    float* o = ee + (size_t)e * DIN;
    for (int t = threadIdx.x; t < DIN; t += 256) {
        float v;
        if (t < NRBF) {
            float a = (d - t * sp) * inv_std;
            v = __expf(-0.5f * a * a);
        } else if (t < NRBF + CCH) {
            v = ese[(size_t)ss * 128 + (t - NRBF)];
        } else {
            v = ese[(size_t)rs * 128 + 64 + (t - NRBF - CCH)];
        }
        o[t] = v;
    }
}

__global__ void k_zero(float* __restrict__ p, long long n) {
    long long i = (long long)blockIdx.x * blockDim.x + threadIdx.x;
    if (i < n) p[i] = 0.0f;
}

__global__ void k_setl0(const float* __restrict__ semb, const int* __restrict__ nspec,
                        float* __restrict__ x, int N) {
    int i = blockIdx.x * blockDim.x + threadIdx.x;
    if (i >= N * CCH) return;
    int n = i >> 6, c = i & 63;
    x[(size_t)n * 1600 + c] = semb[(size_t)nspec[n] * CCH + c];
}

__global__ void k_initmaxden(unsigned* __restrict__ nmax, float* __restrict__ nden, int n) {
    int i = blockIdx.x * blockDim.x + threadIdx.x;
    if (i < n) { nmax[i] = fenc(-3e38f); nden[i] = 0.0f; }
}

__global__ void k_rmsnorm(const float* __restrict__ x, const float* __restrict__ w,
                          float* __restrict__ xn, int N) {
    int n = blockIdx.x;
    int c = threadIdx.x;            // blockDim = 64
    const float* xr = x + (size_t)n * 1600 + c;
    float ss = 0.0f;
    #pragma unroll
    for (int l = 0; l < 25; ++l) { float v = xr[l * 64]; ss += v * v; }
    float inv = rsqrtf(ss * (1.0f / 25.0f) + 1e-6f);
    float* o = xn + (size_t)n * 1600 + c;
    #pragma unroll
    for (int l = 0; l < 25; ++l) o[l * 64] = xr[l * 64] * inv * w[c_LOC[l] * 64 + c];
}

// msg[e,m,0:128] = sum_l wigner[e,m,l] * concat(xn[snd,l,:], xn[rcv,l,:])
__global__ __launch_bounds__(256)
void k_msg_rot(const float* __restrict__ xn, const float* __restrict__ wig,
               const int* __restrict__ snd, const int* __restrict__ rcv,
               float* __restrict__ msg, int E)
{
    int e = blockIdx.x;
    if (e >= E) return;
    __shared__ float sx[1600], sr[1600], sw[475];
    int s = snd[e], rr = rcv[e];
    lds_copy_async(sx, xn + (size_t)s  * 1600, 1600);
    lds_copy_async(sr, xn + (size_t)rr * 1600, 1600);
    for (int t = threadIdx.x; t < 475; t += 256) sw[t] = wig[(size_t)e * 475 + t];
    __syncthreads();
    for (int t = threadIdx.x; t < M2C * 128; t += 256) {
        int m = t >> 7, c = t & 127;
        const float* col = (c < 64) ? (sx + c) : (sr + (c - 64));
        const float* w = sw + m * 25;
        float acc = 0.0f;
        #pragma unroll
        for (int l = 0; l < 25; ++l) acc += w[l] * col[l * 64];
        msg[(size_t)e * 2432 + t] = acc;
    }
}

__global__ void k_modulate(float* __restrict__ msg, const float* __restrict__ rad, long long total) {
    long long i = (long long)blockIdx.x * blockDim.x + threadIdx.x;
    if (i >= total) return;
    long long e = i / 2432;
    int t = (int)(i - e * 2432);
    int m = t >> 7, c = t & 127;
    msg[i] *= rad[e * 384 + c_MG[m] * 128 + c];
}

__global__ __launch_bounds__(256)
void k_logits(const float* __restrict__ aA, const float* __restrict__ avec,
              const int* __restrict__ rcv, float* __restrict__ logits,
              unsigned* __restrict__ nmax, int E)
{
    int e = blockIdx.x;
    if (e >= E) return;
    int lane = threadIdx.x & 31;
    int head = threadIdx.x >> 5;
    float t = silu_f(aA[(size_t)e * 256 + threadIdx.x]) * avec[head * 32 + lane];
    #pragma unroll
    for (int off = 16; off > 0; off >>= 1) t += __shfl_xor(t, off, 32);
    if (lane == 0) {
        logits[(size_t)e * HEADS + head] = t;
        atomicMax(&nmax[(size_t)rcv[e] * HEADS + head], fenc(t));
    }
}

__global__ void k_expden(float* __restrict__ logits, const unsigned* __restrict__ nmax,
                         float* __restrict__ nden, const int* __restrict__ rcv, int E)
{
    int i = blockIdx.x * blockDim.x + threadIdx.x;
    if (i >= E * HEADS) return;
    int e = i >> 3, h = i & 7;
    float mx = fdec(nmax[(size_t)rcv[e] * HEADS + h]);
    float ex = __expf(logits[i] - mx);
    logits[i] = ex;
    atomicAdd(&nden[(size_t)rcv[e] * HEADS + h], ex);
}

__global__ void k_scale_attn(float* __restrict__ v, const float* __restrict__ ex,
                             const float* __restrict__ nden, const int* __restrict__ rcv,
                             long long total)
{
    long long i = (long long)blockIdx.x * blockDim.x + threadIdx.x;
    if (i >= total) return;
    long long e = i / 2432;
    int t = (int)(i - e * 2432);
    int c = t & 127;
    int h = c >> 4;                 // VCH = 16
    float attn = ex[e * HEADS + h] / (nden[(size_t)rcv[e] * HEADS + h] + 1e-12f);
    v[i] *= attn;
}

// x[rcv, l, c] += scale * sum_m wigner_inv[e,l,m] * src[e,m,c]
__global__ __launch_bounds__(256)
void k_rot_scatter(const float* __restrict__ src, const float* __restrict__ winv,
                   const int* __restrict__ rcv, float* __restrict__ x, float scale, int E)
{
    int e = blockIdx.x;
    if (e >= E) return;
    __shared__ float sh[M2C * 64], sw[475];
    lds_copy_async(sh, src + (size_t)e * 1216, 1216);
    for (int t = threadIdx.x; t < 475; t += 256) sw[t] = winv[(size_t)e * 475 + t];
    __syncthreads();
    float* xr = x + (size_t)rcv[e] * 1600;
    for (int t = threadIdx.x; t < 1600; t += 256) {
        int l = t >> 6, c = t & 63;
        const float* w = sw + l * 19;
        float acc = 0.0f;
        #pragma unroll
        for (int m = 0; m < 19; ++m) acc += w[m] * sh[m * 64 + c];
        atomicAdd(xr + t, acc * scale);
    }
}

__global__ __launch_bounds__(256)
void k_togrid(const float* __restrict__ xn, const float* __restrict__ tg,
              float* __restrict__ g, int N)
{
    int n = blockIdx.x;
    __shared__ float sx[1600];
    lds_copy_async(sx, xn + (size_t)n * 1600, 1600);
    __syncthreads();
    for (int t = threadIdx.x; t < GRD * 64; t += 256) {
        int gp = t >> 6, c = t & 63;
        const float* w = tg + gp * 25;
        float acc = 0.0f;
        #pragma unroll
        for (int l = 0; l < 25; ++l) acc += w[l] * sx[l * 64 + c];
        g[(size_t)n * (GRD * 64) + t] = acc;
    }
}

// x[n,l,c] += sum_gp from_grid[l,gp] * g[n,gp,c]   (chunked LDS staging)
__global__ __launch_bounds__(256)
void k_fromgrid_acc(const float* __restrict__ g, const float* __restrict__ fg,
                    float* __restrict__ x, int N)
{
    int n = blockIdx.x;
    __shared__ float sg[96 * 64];
    float acc[7];
    #pragma unroll
    for (int j = 0; j < 7; ++j) acc[j] = 0.0f;
    for (int ch = 0; ch < 3; ++ch) {
        lds_copy_async(sg, g + (size_t)n * (GRD * 64) + (size_t)ch * (96 * 64), 96 * 64);
        __syncthreads();
        #pragma unroll
        for (int j = 0; j < 7; ++j) {
            int t = threadIdx.x + j * 256;
            if (t < 1600) {
                int l = t >> 6, c = t & 63;
                const float* w = fg + l * GRD + ch * 96;
                float a = 0.0f;
                #pragma unroll 8
                for (int gp = 0; gp < 96; ++gp) a += w[gp] * sg[gp * 64 + c];
                acc[j] += a;
            }
        }
        __syncthreads();
    }
    #pragma unroll
    for (int j = 0; j < 7; ++j) {
        int t = threadIdx.x + j * 256;
        if (t < 1600) x[(size_t)n * 1600 + t] += acc[j];
    }
}

// out[n] = (sum_gp from_grid[0,gp] * (hE[n,gp,:] . en_W2 + en_b2)) / AVG_NODES
__global__ __launch_bounds__(256)
void k_energy(const float* __restrict__ hE, const float* __restrict__ enW2,
              const float* __restrict__ enb2, const float* __restrict__ fg,
              float* __restrict__ out, int N)
{
    int n = blockIdx.x;
    __shared__ float red[256];
    float b2 = enb2[0];
    float partial = 0.0f;
    for (int gp = threadIdx.x; gp < GRD; gp += 256) {
        const float* hrow = hE + ((size_t)n * GRD + gp) * HFF;
        float t = b2;
        #pragma unroll 16
        for (int k = 0; k < HFF; ++k) t += hrow[k] * enW2[k];
        partial += fg[gp] * t;
    }
    red[threadIdx.x] = partial;
    __syncthreads();
    for (int s = 128; s > 0; s >>= 1) {
        if (threadIdx.x < s) red[threadIdx.x] += red[threadIdx.x + s];
        __syncthreads();
    }
    if (threadIdx.x == 0) out[n] = red[0] * (1.0f / 2000.0f);
}

// ---------------------------------------------------------------------------
// Host launch
// ---------------------------------------------------------------------------
static const int h_MG[19] = {0, 1,0,1, 2,1,0,1,2, 2,1,0,1,2, 2,1,0,1,2};

static inline dim3 gemm_grid(int M, int N) { return dim3((M + 127) / 128, (N + 63) / 64); }

extern "C" void kernel_launch(void* const* d_in, const int* in_sizes, int n_in,
                              void* d_out, int out_size, void* d_ws, size_t ws_size,
                              hipStream_t stream)
{
    (void)n_in; (void)out_size; (void)ws_size;
    const float* ev      = (const float*)d_in[0];
    const float* wig     = (const float*)d_in[1];
    const float* winv    = (const float*)d_in[2];
    const float* semb    = (const float*)d_in[3];
    const float* ese     = (const float*)d_in[4];
    const float* deg_W1  = (const float*)d_in[5];
    const float* deg_b1  = (const float*)d_in[6];
    const float* deg_W2  = (const float*)d_in[7];
    const float* deg_b2  = (const float*)d_in[8];
    const float* norm1_w = (const float*)d_in[9];
    const float* rad_W1  = (const float*)d_in[10];
    const float* rad_b1  = (const float*)d_in[11];
    const float* rad_W2  = (const float*)d_in[12];
    const float* rad_b2  = (const float*)d_in[13];
    const float* so2_W   = (const float*)d_in[14];
    const float* alpha_W = (const float*)d_in[15];
    const float* alpha_v = (const float*)d_in[16];
    const float* value_W = (const float*)d_in[17];
    const float* proj_W  = (const float*)d_in[18];
    const float* norm2_w = (const float*)d_in[19];
    const float* ffn_W1  = (const float*)d_in[20];
    const float* ffn_b1  = (const float*)d_in[21];
    const float* ffn_W2  = (const float*)d_in[22];
    const float* to_grid = (const float*)d_in[23];
    const float* fr_grid = (const float*)d_in[24];
    const float* fin_w   = (const float*)d_in[25];
    const float* en_W1   = (const float*)d_in[26];
    const float* en_b1   = (const float*)d_in[27];
    const float* en_W2   = (const float*)d_in[28];
    const float* en_b2   = (const float*)d_in[29];
    const int*   nspec   = (const int*)d_in[30];
    const int*   snd     = (const int*)d_in[31];
    const int*   rcv     = (const int*)d_in[32];

    const int N = in_sizes[30];
    const int E = in_sizes[31];
    const int NG = N * GRD;
    float* out = (float*)d_out;

    // workspace carve-up
    float* ws   = (float*)d_ws;
    float* ee   = ws; ws += (size_t)E * DIN;
    float* h1   = ws; ws += (size_t)E * 128;
    float* big1 = ws; ws += (size_t)E * 2432;   // msg / v / ffn out
    float* big2 = ws; ws += (size_t)E * 1216;   // deg h / so2 h
    float* big3 = ws; ws += (size_t)E * 1216;   // oe
    float* radb = ws; ws += (size_t)E * 384;
    float* aA   = ws; ws += (size_t)E * 256;
    float* exb  = ws; ws += (size_t)E * HEADS;
    float* x    = ws; ws += (size_t)N * 1600;
    float* xn   = ws; ws += (size_t)N * 1600;
    float* g1   = ws; ws += (size_t)NG * 64;
    float* g2   = ws; ws += (size_t)NG * 128;
    unsigned* nmax = (unsigned*)ws; ws += (size_t)N * HEADS;
    float* nden = ws; ws += (size_t)N * HEADS;

    // 1) edge features
    k_ee<<<E, 256, 0, stream>>>(ev, ese, snd, rcv, nspec, ee, E);

    // 2) node init
    {
        long long nx = (long long)N * 1600;
        k_zero<<<(unsigned)((nx + 255) / 256), 256, 0, stream>>>(x, nx);
        k_setl0<<<(N * CCH + 255) / 256, 256, 0, stream>>>(semb, nspec, x, N);
    }

    // 3) edge-degree embedding MLP + rotate-back + scatter
    k_gemm<0, 1><<<gemm_grid(E, 128), 256, 0, stream>>>(ee, DIN, deg_W1, 128, deg_b1, h1, 128, E, 128, DIN);
    k_gemm<0, 0><<<gemm_grid(E, 1216), 256, 0, stream>>>(h1, 128, deg_W2, 1216, deg_b2, big2, 1216, E, 1216, 128);
    k_rot_scatter<<<E, 256, 0, stream>>>(big2, winv, rcv, x, 1.0f / 20.0f, E);

    for (int i = 0; i < NLAY; ++i) {
        // ---- attention ----
        k_rmsnorm<<<N, 64, 0, stream>>>(x, norm1_w + i * 320, xn, N);
        k_msg_rot<<<E, 256, 0, stream>>>(xn, wig, snd, rcv, big1, E);

        k_gemm<0, 1><<<gemm_grid(E, 128), 256, 0, stream>>>(ee, DIN, rad_W1 + (size_t)i * DIN * 128, 128,
                                                            rad_b1 + i * 128, h1, 128, E, 128, DIN);
        k_gemm<0, 0><<<gemm_grid(E, 384), 256, 0, stream>>>(h1, 128, rad_W2 + (size_t)i * 128 * 384, 384,
                                                            rad_b2 + i * 384, radb, 384, E, 384, 128);
        {
            long long tot = (long long)E * 2432;
            k_modulate<<<(unsigned)((tot + 255) / 256), 256, 0, stream>>>(big1, radb, tot);
        }
        // SO2 conv: 19 slot-GEMMs with per-|m| weights
        for (int s = 0; s < 19; ++s)
            k_gemm<0, 0><<<gemm_grid(E, HMSG), 256, 0, stream>>>(
                big1 + s * 128, 2432,
                so2_W + (size_t)i * 3 * 128 * 64 + (size_t)h_MG[s] * 128 * 64, 64,
                nullptr, big2 + s * 64, 1216, E, HMSG, 128);
        // alpha path (silu applied to h[:,0,:] on input)
        k_gemm<1, 0><<<gemm_grid(E, 256), 256, 0, stream>>>(big2, 1216,
                                                            alpha_W + (size_t)i * 64 * 256, 256,
                                                            nullptr, aA, 256, E, 256, 64);
        k_initmaxden<<<(N * HEADS + 255) / 256, 256, 0, stream>>>(nmax, nden, N * HEADS);
        k_logits<<<E, 256, 0, stream>>>(aA, alpha_v + i * 256, rcv, exb, nmax, E);
        k_expden<<<(E * HEADS + 255) / 256, 256, 0, stream>>>(exb, nmax, nden, rcv, E);
        // value
        for (int s = 0; s < 19; ++s)
            k_gemm<0, 0><<<gemm_grid(E, 128), 256, 0, stream>>>(
                big2 + s * 64, 1216,
                value_W + (size_t)i * 3 * 64 * 128 + (size_t)h_MG[s] * 64 * 128, 128,
                nullptr, big1 + s * 128, 2432, E, 128, 64);
        {
            long long tot = (long long)E * 2432;
            k_scale_attn<<<(unsigned)((tot + 255) / 256), 256, 0, stream>>>(big1, exb, nden, rcv, tot);
        }
        // projection
        for (int s = 0; s < 19; ++s)
            k_gemm<0, 0><<<gemm_grid(E, 64), 256, 0, stream>>>(
                big1 + s * 128, 2432,
                proj_W + (size_t)i * 3 * 128 * 64 + (size_t)h_MG[s] * 128 * 64, 64,
                nullptr, big3 + s * 64, 1216, E, 64, 128);
        k_rot_scatter<<<E, 256, 0, stream>>>(big3, winv, rcv, x, 1.0f, E);

        // ---- S2 grid FFN ----
        k_rmsnorm<<<N, 64, 0, stream>>>(x, norm2_w + i * 320, xn, N);
        k_togrid<<<N, 256, 0, stream>>>(xn, to_grid, g1, N);
        k_gemm<0, 1><<<gemm_grid(NG, HFF), 256, 0, stream>>>(g1, 64, ffn_W1 + (size_t)i * 64 * 128, 128,
                                                             ffn_b1 + i * 128, g2, 128, NG, HFF, 64);
        k_gemm<0, 0><<<gemm_grid(NG, 64), 256, 0, stream>>>(g2, 128, ffn_W2 + (size_t)i * 128 * 64, 64,
                                                            nullptr, big1, 64, NG, 64, HFF);
        k_fromgrid_acc<<<N, 256, 0, stream>>>(big1, fr_grid, x, N);
    }

    // final norm + energy head
    k_rmsnorm<<<N, 64, 0, stream>>>(x, fin_w, xn, N);
    k_togrid<<<N, 256, 0, stream>>>(xn, to_grid, g1, N);
    k_gemm<0, 1><<<gemm_grid(NG, HFF), 256, 0, stream>>>(g1, 64, en_W1, 128, en_b1, g2, 128, NG, HFF, 64);
    k_energy<<<N, 256, 0, stream>>>(g2, en_W2, en_b2, fr_grid, out, N);
}
